// GCNConv_62328565400241
// MI455X (gfx1250) — matrix operand order repaired and checked by reading, not compiled, not run on hardware
//
#include <hip/hip_runtime.h>

typedef __attribute__((ext_vector_type(16))) __bf16 v16bf;
typedef __attribute__((ext_vector_type(8)))  float  v8f;

#define F 32  // F_IN == F_OUT == 32

// ---------------------------------------------------------------------------
// Kernel 1: zero the output accumulator (128-bit stores, grid-stride)
// ---------------------------------------------------------------------------
__global__ void zero_f4(float4* __restrict__ p, long n4) {
  long i = (long)blockIdx.x * blockDim.x + threadIdx.x;
  long stride = (long)gridDim.x * blockDim.x;
  float4 z; z.x = 0.f; z.y = 0.f; z.z = 0.f; z.w = 0.f;
  for (; i < n4; i += stride) p[i] = z;
}

// ---------------------------------------------------------------------------
// Kernel 2: y[N,32] = x[N,32] @ W[32,32]  via v_wmma_f32_16x16x32_bf16
// One wave handles one 16-row tile; two WMMAs cover the 32 output columns.
// A layout (16-bit A 16x32): lane<16 -> M=lane,    K = {0..7, 16..23}
//                            lane>=16 -> M=lane-16, K = {8..15, 24..31}
// B layout (16-bit B 32x16): lane<16 -> K = {0..7, 16..23},  N = lane
//                            lane>=16 -> K = {8..15, 24..31}, N = lane-16
// D layout (f32 16x16): VGPR v -> M = v + 8*(lane>=16), N = lane&15
// ---------------------------------------------------------------------------
__global__ void xw_wmma_bf16(const float* __restrict__ x,
                             const float* __restrict__ w,
                             float* __restrict__ y,
                             int n_nodes) {
  const int lane = threadIdx.x & 31;
  const int tile = blockIdx.x * (blockDim.x >> 5) + (threadIdx.x >> 5);
  const int row0 = tile << 4;
  if (row0 >= n_nodes) return;          // wave-uniform exit: EXEC stays all-1s for WMMA

  const int m  = lane & 15;
  const int hi = lane >> 4;             // 0 = lanes 0-15, 1 = lanes 16-31

  // ---- A tile: clamp row for the (possible) partial tail tile ----
  int arow = row0 + m;
  if (arow > n_nodes - 1) arow = n_nodes - 1;
  const float4* xr = (const float4*)(x + (size_t)arow * F);
  const int kq = hi * 2;                // K-quarter offset in float4 units
  float4 q0 = xr[kq];                   // K = 8*hi + 0..3
  float4 q1 = xr[kq + 1];               // K = 8*hi + 4..7
  float4 q2 = xr[kq + 4];               // K = 16 + 8*hi + 0..3
  float4 q3 = xr[kq + 5];               // K = 16 + 8*hi + 4..7
  v16bf a;
  a[0]  = (__bf16)q0.x;  a[1]  = (__bf16)q0.y;  a[2]  = (__bf16)q0.z;  a[3]  = (__bf16)q0.w;
  a[4]  = (__bf16)q1.x;  a[5]  = (__bf16)q1.y;  a[6]  = (__bf16)q1.z;  a[7]  = (__bf16)q1.w;
  a[8]  = (__bf16)q2.x;  a[9]  = (__bf16)q2.y;  a[10] = (__bf16)q2.z;  a[11] = (__bf16)q2.w;
  a[12] = (__bf16)q3.x;  a[13] = (__bf16)q3.y;  a[14] = (__bf16)q3.z;  a[15] = (__bf16)q3.w;

  // ---- B tiles: W is 4 KB, L1-resident; loads are broadcast-friendly ----
  const int kb = hi << 3;
  v16bf b0, b1;
#pragma unroll
  for (int i = 0; i < 16; ++i) {
    b0[i] = (__bf16)w[(kb + i + ((i >> 3) << 3)) * F + m];        // columns 0..15
    b1[i] = (__bf16)w[(kb + i + ((i >> 3) << 3)) * F + 16 + m];   // columns 16..31
  }

  v8f c0 = {}, c1 = {};
  c0 = __builtin_amdgcn_wmma_f32_16x16x32_bf16(false, a, false, b0, (short)0, c0, false, false);
  c1 = __builtin_amdgcn_wmma_f32_16x16x32_bf16(false, a, false, b1, (short)0, c1, false, false);

  // ---- D store: single wave-uniform range check, then an unguarded clause
  //      of immediate-offset b32 stores from one base pointer. ----
  float* yb = y + (size_t)(row0 + (hi << 3)) * F + m;   // row = row0 + 8*hi, col = m
  if (row0 + 16 <= n_nodes) {
#pragma unroll
    for (int v = 0; v < 8; ++v) {
      yb[v * F]      = c0[v];
      yb[v * F + 16] = c1[v];
    }
  } else {
#pragma unroll
    for (int v = 0; v < 8; ++v) {
      if (row0 + (hi << 3) + v < n_nodes) {
        yb[v * F]      = c0[v];
        yb[v * F + 16] = c1[v];
      }
    }
  }
}

// ---------------------------------------------------------------------------
// Kernel 3: out[row] += val * y[col]   (rows sorted -> run-length accumulate,
// flush at segment boundaries with native f32 atomics). lane = feature.
// Edge metadata loaded 32-wide and broadcast via __shfl (ds_bpermute).
// ---------------------------------------------------------------------------
__global__ void edge_agg(const int* __restrict__ rows,
                         const int* __restrict__ cols,
                         const float* __restrict__ vals,
                         const float* __restrict__ y,
                         float* __restrict__ out,
                         long n_edges, int edges_per_wave) {
  const int lane = threadIdx.x & 31;
  const long wv = ((long)blockIdx.x * blockDim.x + threadIdx.x) >> 5;
  long e0 = wv * (long)edges_per_wave;
  if (e0 >= n_edges) return;
  long e1 = e0 + edges_per_wave;
  if (e1 > n_edges) e1 = n_edges;

  float acc = 0.f;
  int   cur = -1;

  for (long base = e0; base < e1; base += 32) {
    long e = base + lane;
    bool ok = e < e1;
    int   r = ok ? rows[e] : 0;
    int   c = ok ? cols[e] : 0;
    float v = ok ? vals[e] : 0.f;

    if (base + 256 < n_edges) {           // stream index arrays ahead of use
      __builtin_prefetch(&rows[base + 256], 0, 1);
      __builtin_prefetch(&cols[base + 256], 0, 1);
      __builtin_prefetch(&vals[base + 256], 0, 1);
    }

    const int cnt = (int)(((e1 - base) < 32) ? (e1 - base) : 32);
    if (cnt == 32) {
#pragma unroll 8
      for (int j = 0; j < 32; ++j) {
        int   rj = __shfl(r, j, 32);
        int   cj = __shfl(c, j, 32);
        float vj = __shfl(v, j, 32);
        if (rj != cur) {                   // wave-uniform branch
          if (cur >= 0) unsafeAtomicAdd(&out[(size_t)cur * F + lane], acc);
          cur = rj; acc = 0.f;
        }
        acc = fmaf(vj, y[(size_t)cj * F + lane], acc);
      }
    } else {
      for (int j = 0; j < cnt; ++j) {
        int   rj = __shfl(r, j, 32);
        int   cj = __shfl(c, j, 32);
        float vj = __shfl(v, j, 32);
        if (rj != cur) {
          if (cur >= 0) unsafeAtomicAdd(&out[(size_t)cur * F + lane], acc);
          cur = rj; acc = 0.f;
        }
        acc = fmaf(vj, y[(size_t)cj * F + lane], acc);
      }
    }
  }
  if (cur >= 0) unsafeAtomicAdd(&out[(size_t)cur * F + lane], acc);
}

// ---------------------------------------------------------------------------
// Launch: zero(out) -> y = x@W (WMMA) -> edge scatter-accumulate into out
// Uses the identity (A x) W == A (x W): dense transform first, so the edge
// gather reads the L2-resident transformed features directly into the output.
// ---------------------------------------------------------------------------
extern "C" void kernel_launch(void* const* d_in, const int* in_sizes, int n_in,
                              void* d_out, int out_size, void* d_ws, size_t ws_size,
                              hipStream_t stream) {
  const float* x    = (const float*)d_in[0];
  const int*   rows = (const int*)d_in[1];
  const int*   cols = (const int*)d_in[2];
  const float* vals = (const float*)d_in[3];
  const float* w    = (const float*)d_in[4];
  float* out = (float*)d_out;

  const int  n_nodes = in_sizes[0] / F;
  const long n_edges = (long)in_sizes[3];

  float* y = (float*)d_ws;   // n_nodes * F floats of scratch (12.8 MB)

  // 1) zero the accumulation target
  long n4 = (long)out_size / 4;
  zero_f4<<<1024, 256, 0, stream>>>((float4*)out, n4);

  // 2) dense feature transform via WMMA (one wave per 16-row tile)
  const int wavesPerBlock = 4;
  int tiles  = (n_nodes + 15) / 16;
  int blocks = (tiles + wavesPerBlock - 1) / wavesPerBlock;
  xw_wmma_bf16<<<blocks, wavesPerBlock * 32, 0, stream>>>(x, w, y, n_nodes);

  // 3) sparse aggregation (CSR-like sorted rows, register run accumulation)
  const int  EPW   = 128;
  const int  wpb   = 8;
  long waves  = (n_edges + EPW - 1) / EPW;
  long nblk   = (waves + wpb - 1) / wpb;
  edge_agg<<<(int)nblk, wpb * 32, 0, stream>>>(rows, cols, vals, y, out, n_edges, EPW);
}
